// TransformerLayer_20048907337977
// MI455X (gfx1250) — matrix operand order repaired and checked by reading, not compiled, not run on hardware
//
#include <hip/hip_runtime.h>
#include <hip/hip_bf16.h>
#include <math.h>

// ---------------------------------------------------------------------------
// Types for CDNA5 WMMA (wave32): D = A(16x32 bf16) * B(32x16 bf16) + C(16x16 f32)
// ---------------------------------------------------------------------------
typedef __attribute__((ext_vector_type(16))) __bf16 v16bf;
typedef __attribute__((ext_vector_type(8)))  __bf16 v8bf;
typedef __attribute__((ext_vector_type(8)))  float  v8f;

__device__ __forceinline__ v8f wmma_bf16(v16bf a, v16bf b, v8f c) {
  return __builtin_amdgcn_wmma_f32_16x16x32_bf16(
      /*neg_a=*/false, a, /*neg_b=*/false, b,
      /*c_mod=*/(short)0, c, /*reuse_a=*/false, /*reuse_b=*/false);
}

// Concatenate two 8-wide bf16 chunks (A-fragment: K = base+0..7 and base+16..23)
__device__ __forceinline__ v16bf afrag_from(const __bf16* p) {
  v8bf lo = *(const v8bf*)(p);
  v8bf hi = *(const v8bf*)(p + 16);
  return __builtin_shufflevector(lo, hi, 0,1,2,3,4,5,6,7,8,9,10,11,12,13,14,15);
}

// ---------------------------------------------------------------------------
// CDNA5 async global->LDS copy (ASYNCcnt-tracked; bypasses VGPRs).
// GVS addressing: 64-bit SGPR base + per-lane 32-bit byte offset.
// vdst VGPR carries the LDS byte address (low 32 bits of the generic pointer).
// ---------------------------------------------------------------------------
__device__ __forceinline__ void async_b128(const void* base, unsigned voff, void* ldsp) {
  unsigned lds = (unsigned)(unsigned long long)ldsp;
  asm volatile("global_load_async_to_lds_b128 %0, %1, %2"
               :: "v"(lds), "v"(voff), "s"(base) : "memory");
}
__device__ __forceinline__ void wait_async0() {
  asm volatile("s_wait_asynccnt 0x0" ::: "memory");
}

static constexpr int Bd = 4, Sd = 2048, Dd = 1024, Hd = 16, Ed = 64;
static constexpr int ROWS = Bd * Sd;   // 8192

// ---------------------------------------------------------------------------
// Conversion / re-layout kernels (one-time, bandwidth-trivial)
// ---------------------------------------------------------------------------
__global__ void cvt_f32_bf16(const float* __restrict__ in, __bf16* __restrict__ out, int n) {
  for (int i = blockIdx.x * blockDim.x + threadIdx.x; i < n; i += gridDim.x * blockDim.x)
    out[i] = (__bf16)in[i];
}

// Fused, transposed QKV weight: out[j][d], j = m*1024 + h*64 + e, m in {q,k,v}
__global__ void cvt_qkv_w(const float* __restrict__ Wq, const float* __restrict__ Wk,
                          const float* __restrict__ Wv, __bf16* __restrict__ out) {
  int j = blockIdx.x;                 // 0..3071
  int m = j >> 10, c = j & 1023;
  int h = c >> 6,  e = c & 63;
  const float* W = (m == 0) ? Wq : (m == 1) ? Wk : Wv;   // [H][D][E]
  const float* src = W + (size_t)h * Dd * Ed + e;
  __bf16* dst = out + (size_t)j * Dd;
  for (int d = threadIdx.x; d < Dd; d += blockDim.x)
    dst[d] = (__bf16)src[(size_t)d * Ed];
}

// Transpose square [K][N] fp32 -> [N][K] bf16
__global__ void cvt_w_t(const float* __restrict__ W, __bf16* __restrict__ out) {
  int n = blockIdx.x;
  for (int k = threadIdx.x; k < Dd; k += blockDim.x)
    out[(size_t)n * Dd + k] = (__bf16)W[(size_t)k * Dd + n];
}

// ---------------------------------------------------------------------------
// Tiled bf16 WMMA GEMM: C[M,N] = A[M,K] * Bt[N,K]^T (+epilogue)
// block = 256 threads (8 waves), tile 128x128, BK=32, async double-buffered LDS.
// wave grid: 4 (M) x 2 (N); each wave owns 2x4 16x16 accumulators.
// MODE 0: QKV scatter + bias -> bf16 [B,H,S,E]
// MODE 1: +bias -> fp32 [M,N]
// MODE 2: +bias, ReLU -> bf16 [M,N]
// ---------------------------------------------------------------------------
template <int MODE>
__global__ __launch_bounds__(256) void gemm_bf16_wmma(
    const __bf16* __restrict__ A, const __bf16* __restrict__ Bt,
    int M, int N, int K,
    const float* __restrict__ bias, float* __restrict__ outF, __bf16* __restrict__ outB,
    __bf16* __restrict__ outQ, __bf16* __restrict__ outK, __bf16* __restrict__ outV,
    const float* __restrict__ bq, const float* __restrict__ bk, const float* __restrict__ bv)
{
  __shared__ alignas(32) __bf16 sA[2][128 * 32];
  __shared__ alignas(32) __bf16 sB[2][128 * 32];

  const int t = threadIdx.x;
  const int lane = t & 31, w = t >> 5;
  const int wm = w & 3, wn = w >> 2;
  const int lm = lane & 15, lh = lane >> 4;
  const int bm = blockIdx.y * 128, bn = blockIdx.x * 128;

  v8f acc[2][4];
  #pragma unroll
  for (int mt = 0; mt < 2; ++mt)
    #pragma unroll
    for (int nt = 0; nt < 4; ++nt)
      acc[mt][nt] = (v8f)0.0f;

  // staging: each thread moves 32B of A and 32B of B per K-tile via async LDS DMA
  const int srow = t >> 1, shalf = t & 1;
  const unsigned aoff0 = ((unsigned)(bm + srow) * (unsigned)K + shalf * 16) * 2u;
  const unsigned boff0 = ((unsigned)(bn + srow) * (unsigned)K + shalf * 16) * 2u;
  const int sidx = srow * 32 + shalf * 16;

  auto stage = [&](int buf, int k0) {
    unsigned ao = aoff0 + (unsigned)k0 * 2u;
    unsigned bo = boff0 + (unsigned)k0 * 2u;
    async_b128(A,  ao,      &sA[buf][sidx]);
    async_b128(A,  ao + 16, &sA[buf][sidx + 8]);
    async_b128(Bt, bo,      &sB[buf][sidx]);
    async_b128(Bt, bo + 16, &sB[buf][sidx + 8]);
  };

  stage(0, 0);
  wait_async0();
  __syncthreads();

  int buf = 0;
  for (int k0 = 0; k0 < K; k0 += 32) {
    if (k0 + 32 < K) stage(buf ^ 1, k0 + 32);   // prefetch next tile (async DMA)

    v16bf af[2];
    #pragma unroll
    for (int mt = 0; mt < 2; ++mt)
      af[mt] = afrag_from(&sA[buf][(wm * 32 + mt * 16 + lm) * 32 + lh * 8]);
    #pragma unroll
    for (int nt = 0; nt < 4; ++nt) {
      v16bf bf_ = afrag_from(&sB[buf][(wn * 64 + nt * 16 + lm) * 32 + lh * 8]);
      #pragma unroll
      for (int mt = 0; mt < 2; ++mt)
        acc[mt][nt] = wmma_bf16(af[mt], bf_, acc[mt][nt]);
    }

    wait_async0();          // next tile landed in LDS
    __syncthreads();        // publish to all waves / protect current tile reads
    buf ^= 1;
  }

  // Epilogue. C layout: element r of lane l -> row r + 8*(l>>4), col l&15
  #pragma unroll
  for (int mt = 0; mt < 2; ++mt)
    #pragma unroll
    for (int nt = 0; nt < 4; ++nt)
      #pragma unroll
      for (int r = 0; r < 8; ++r) {
        int row = bm + wm * 32 + mt * 16 + r + 8 * lh;
        int col = bn + wn * 64 + nt * 16 + lm;
        float v = acc[mt][nt][r];
        if (MODE == 0) {
          int m = col >> 10, c = col & 1023;
          int h = c >> 6, e = c & 63;
          int b = row >> 11, s = row & (Sd - 1);
          const float* bp = (m == 0) ? bq : (m == 1) ? bk : bv;
          __bf16* dst = (m == 0) ? outQ : (m == 1) ? outK : outV;
          dst[(((size_t)(b * Hd + h)) * Sd + s) * Ed + e] = (__bf16)(v + bp[c]);
        } else if (MODE == 1) {
          outF[(size_t)row * N + col] = v + bias[col];
        } else {
          float y = v + bias[col];
          outB[(size_t)row * N + col] = (__bf16)fmaxf(y, 0.0f);
        }
      }
}

// ---------------------------------------------------------------------------
// Flash attention: one block per (b, h, 64 query rows); 4 waves x 16 rows.
// Q/K tiles staged via async LDS DMA; V transposed through VGPRs; online softmax.
// ---------------------------------------------------------------------------
__global__ __launch_bounds__(128) void attention_wmma(
    const __bf16* __restrict__ qb, const __bf16* __restrict__ kb,
    const __bf16* __restrict__ vb, __bf16* __restrict__ ctxb)
{
  __shared__ alignas(32) __bf16 sQ [64 * 64];
  __shared__ alignas(32) __bf16 sK [64 * 64];
  __shared__ alignas(32) __bf16 sVt[64 * 64];          // transposed: [e][s]
  __shared__ alignas(32) __bf16 sP [4 * 16 * 64];      // per-wave P tile

  const int blk = blockIdx.x;
  const int qt = blk & 31;
  const int h  = (blk >> 5) & 15;
  const int b  = blk >> 9;
  const size_t headoff = ((size_t)(b * Hd + h)) * Sd * Ed;
  const int qbase = qt * 64;

  const int t = threadIdx.x, lane = t & 31, w = t >> 5;
  const int lm = lane & 15, lh = lane >> 4;
  const int row = t >> 1, half = t & 1;     // staging coordinates (64B per thread)

  {  // async-load Q tile (64x64 bf16); completion folded into first K-tile wait
    unsigned qoff = (unsigned)((headoff + (size_t)(qbase + row) * Ed + half * 32) * 2);
    __bf16* d = &sQ[row * 64 + half * 32];
    async_b128(qb, qoff,      d);
    async_b128(qb, qoff + 16, d + 8);
    async_b128(qb, qoff + 32, d + 16);
    async_b128(qb, qoff + 48, d + 24);
  }

  float mi[8], li[8];
  v8f o[4];
  #pragma unroll
  for (int r = 0; r < 8; ++r) { mi[r] = -1e30f; li[r] = 0.0f; }
  #pragma unroll
  for (int ne = 0; ne < 4; ++ne) o[ne] = (v8f)0.0f;

  for (int kt = 0; kt < Sd; kt += 64) {
    __syncthreads();   // prior tile's WMMA reads complete before overwrite
    {  // stage K via async DMA
      unsigned koff = (unsigned)((headoff + (size_t)(kt + row) * Ed + half * 32) * 2);
      __bf16* d = &sK[row * 64 + half * 32];
      async_b128(kb, koff,      d);
      async_b128(kb, koff + 16, d + 8);
      async_b128(kb, koff + 32, d + 16);
      async_b128(kb, koff + 48, d + 24);
    }
    {  // stage V transposed ([e][s]) through VGPRs
      const __bf16* gv = vb + headoff + (size_t)(kt + row) * Ed + half * 32;
      v16bf t0 = *(const v16bf*)(gv);
      v16bf t1 = *(const v16bf*)(gv + 16);
      #pragma unroll
      for (int i = 0; i < 16; ++i) sVt[(half * 32 + i) * 64 + row] = t0[i];
      #pragma unroll
      for (int i = 0; i < 16; ++i) sVt[(half * 32 + 16 + i) * 64 + row] = t1[i];
    }
    wait_async0();
    __syncthreads();

    // scores = Q * K^T  (contraction over E=64: two WMMA k-steps)
    v8f sacc[4];
    #pragma unroll
    for (int nt = 0; nt < 4; ++nt) sacc[nt] = (v8f)0.0f;
    #pragma unroll
    for (int ks = 0; ks < 64; ks += 32) {
      v16bf af = afrag_from(&sQ[(w * 16 + lm) * 64 + ks + lh * 8]);
      #pragma unroll
      for (int nt = 0; nt < 4; ++nt) {
        v16bf bf_ = afrag_from(&sK[(nt * 16 + lm) * 64 + ks + lh * 8]);
        sacc[nt] = wmma_bf16(af, bf_, sacc[nt]);
      }
    }

    // online softmax (scale by 1/sqrt(E) = 0.125)
    float ps[4][8];
    #pragma unroll
    for (int r = 0; r < 8; ++r) {
      float mx = -1e30f;
      #pragma unroll
      for (int nt = 0; nt < 4; ++nt) {
        float v = sacc[nt][r] * 0.125f;
        ps[nt][r] = v;
        mx = fmaxf(mx, v);
      }
      mx = fmaxf(mx, __shfl_xor(mx, 1));
      mx = fmaxf(mx, __shfl_xor(mx, 2));
      mx = fmaxf(mx, __shfl_xor(mx, 4));
      mx = fmaxf(mx, __shfl_xor(mx, 8));
      float mn = fmaxf(mi[r], mx);
      float sc = __expf(mi[r] - mn);
      mi[r] = mn;
      float rs = 0.0f;
      #pragma unroll
      for (int nt = 0; nt < 4; ++nt) {
        float p = __expf(ps[nt][r] - mn);
        ps[nt][r] = p;
        rs += p;
      }
      rs += __shfl_xor(rs, 1);
      rs += __shfl_xor(rs, 2);
      rs += __shfl_xor(rs, 4);
      rs += __shfl_xor(rs, 8);
      li[r] = li[r] * sc + rs;
      #pragma unroll
      for (int ne = 0; ne < 4; ++ne) o[ne][r] *= sc;
    }

    // C-layout -> A-fragment layout via per-wave LDS region
    #pragma unroll
    for (int nt = 0; nt < 4; ++nt)
      #pragma unroll
      for (int r = 0; r < 8; ++r)
        sP[w * 1024 + (r + 8 * lh) * 64 + nt * 16 + lm] = (__bf16)ps[nt][r];

    // context += P * V   (contraction over 64 keys: two WMMA k-steps)
    #pragma unroll
    for (int ks = 0; ks < 64; ks += 32) {
      v16bf af = afrag_from(&sP[w * 1024 + lm * 64 + ks + lh * 8]);
      #pragma unroll
      for (int ne = 0; ne < 4; ++ne) {
        v16bf bf_ = afrag_from(&sVt[(ne * 16 + lm) * 64 + ks + lh * 8]);
        o[ne] = wmma_bf16(af, bf_, o[ne]);
      }
    }
  }

  // normalize and write concat layout [B,S,D] (col = h*64 + e), bf16
  #pragma unroll
  for (int ne = 0; ne < 4; ++ne)
    #pragma unroll
    for (int r = 0; r < 8; ++r) {
      int s = qbase + w * 16 + r + 8 * lh;
      int col = h * Ed + ne * 16 + lm;
      float val = o[ne][r] / li[r];
      ctxb[((size_t)(b * Sd + s)) * Dd + col] = (__bf16)val;
    }
}

// ---------------------------------------------------------------------------
// Residual + LayerNorm over D=1024; optionally also emit bf16 copy.
// ---------------------------------------------------------------------------
__global__ __launch_bounds__(256) void ln_residual(
    const float* __restrict__ a, const float* __restrict__ b,
    const float* __restrict__ gamma, const float* __restrict__ beta,
    float* __restrict__ outF, __bf16* __restrict__ outB)
{
  __shared__ float red[256];
  const int row = blockIdx.x, tid = threadIdx.x;
  const float* pa = a + (size_t)row * Dd;
  const float* pb = b + (size_t)row * Dd;

  float loc[4], s = 0.0f;
  #pragma unroll
  for (int i = 0; i < 4; ++i) {
    int j = tid + i * 256;
    loc[i] = pa[j] + pb[j];
    s += loc[i];
  }
  red[tid] = s;
  __syncthreads();
  for (int off = 128; off > 0; off >>= 1) {
    if (tid < off) red[tid] += red[tid + off];
    __syncthreads();
  }
  float mu = red[0] * (1.0f / Dd);
  __syncthreads();

  float vs = 0.0f;
  #pragma unroll
  for (int i = 0; i < 4; ++i) {
    float d = loc[i] - mu;
    vs += d * d;
  }
  red[tid] = vs;
  __syncthreads();
  for (int off = 128; off > 0; off >>= 1) {
    if (tid < off) red[tid] += red[tid + off];
    __syncthreads();
  }
  float inv = rsqrtf(red[0] * (1.0f / Dd) + 1e-6f);

  #pragma unroll
  for (int i = 0; i < 4; ++i) {
    int j = tid + i * 256;
    float y = (loc[i] - mu) * inv * gamma[j] + beta[j];
    outF[(size_t)row * Dd + j] = y;
    if (outB) outB[(size_t)row * Dd + j] = (__bf16)y;
  }
}

// ---------------------------------------------------------------------------
// Host-side orchestration
// ---------------------------------------------------------------------------
extern "C" void kernel_launch(void* const* d_in, const int* in_sizes, int n_in,
                              void* d_out, int out_size, void* d_ws, size_t ws_size,
                              hipStream_t stream) {
  const float* x   = (const float*)d_in[0];
  const float* Wq  = (const float*)d_in[1];
  const float* bq  = (const float*)d_in[2];
  const float* Wk  = (const float*)d_in[3];
  const float* bk  = (const float*)d_in[4];
  const float* Wv  = (const float*)d_in[5];
  const float* bv  = (const float*)d_in[6];
  const float* Wo  = (const float*)d_in[7];
  const float* bo  = (const float*)d_in[8];
  const float* g1  = (const float*)d_in[9];
  const float* be1 = (const float*)d_in[10];
  const float* W1  = (const float*)d_in[11];
  const float* b1  = (const float*)d_in[12];
  const float* W2  = (const float*)d_in[13];
  const float* b2  = (const float*)d_in[14];
  const float* g2  = (const float*)d_in[15];
  const float* be2 = (const float*)d_in[16];
  float* out = (float*)d_out;

  char* p = (char*)d_ws;
  auto carve = [&](size_t bytes) -> char* {
    char* r = p;
    p += (bytes + 255) & ~(size_t)255;
    return r;
  };
  const size_t R = ROWS;  // 8192
  __bf16* xb   = (__bf16*)carve(R * Dd * 2);
  __bf16* wqkv = (__bf16*)carve((size_t)3072 * Dd * 2);
  __bf16* wo_t = (__bf16*)carve((size_t)Dd * Dd * 2);
  __bf16* w1_t = (__bf16*)carve((size_t)Dd * Dd * 2);
  __bf16* w2_t = (__bf16*)carve((size_t)Dd * Dd * 2);
  __bf16* qbuf = (__bf16*)carve(R * Dd * 2);
  __bf16* kbuf = (__bf16*)carve(R * Dd * 2);
  __bf16* vbuf = (__bf16*)carve(R * Dd * 2);
  __bf16* ctxb = (__bf16*)carve(R * Dd * 2);
  __bf16* hb   = (__bf16*)carve(R * Dd * 2);
  __bf16* ffn1 = (__bf16*)carve(R * Dd * 2);
  float*  attn = (float*)carve(R * Dd * 4);
  float*  hf   = (float*)carve(R * Dd * 4);
  float*  ffn2 = (float*)carve(R * Dd * 4);

  // 1) precision conversion / weight re-layout
  cvt_f32_bf16<<<1024, 256, 0, stream>>>(x, xb, (int)(R * Dd));
  cvt_qkv_w<<<3072, 256, 0, stream>>>(Wq, Wk, Wv, wqkv);
  cvt_w_t<<<Dd, 256, 0, stream>>>(Wo, wo_t);
  cvt_w_t<<<Dd, 256, 0, stream>>>(W1, w1_t);
  cvt_w_t<<<Dd, 256, 0, stream>>>(W2, w2_t);

  // 2) fused QKV projection: [8192,1024] x [1024,3072]
  dim3 gq(3072 / 128, (unsigned)(R / 128));
  gemm_bf16_wmma<0><<<gq, 256, 0, stream>>>(xb, wqkv, (int)R, 3072, Dd,
      nullptr, nullptr, nullptr, qbuf, kbuf, vbuf, bq, bk, bv);

  // 3) attention (flash, per 64-row query tile)
  attention_wmma<<<Bd * Hd * (Sd / 64), 128, 0, stream>>>(qbuf, kbuf, vbuf, ctxb);

  // 4) output projection -> attn_out (fp32)
  dim3 go(Dd / 128, (unsigned)(R / 128));
  gemm_bf16_wmma<1><<<go, 256, 0, stream>>>(ctxb, wo_t, (int)R, Dd, Dd,
      bo, attn, nullptr, nullptr, nullptr, nullptr, nullptr, nullptr, nullptr);

  // 5) h = LN(x + attn_out); also bf16 copy for FFN GEMM
  ln_residual<<<(unsigned)R, 256, 0, stream>>>(x, attn, g1, be1, hf, hb);

  // 6) FFN: relu(h W1 + b1) W2 + b2
  gemm_bf16_wmma<2><<<go, 256, 0, stream>>>(hb, w1_t, (int)R, Dd, Dd,
      b1, nullptr, ffn1, nullptr, nullptr, nullptr, nullptr, nullptr, nullptr);
  gemm_bf16_wmma<1><<<go, 256, 0, stream>>>(ffn1, w2_t, (int)R, Dd, Dd,
      b2, ffn2, nullptr, nullptr, nullptr, nullptr, nullptr, nullptr, nullptr);

  // 7) out = LN(h + ffn)
  ln_residual<<<(unsigned)R, 256, 0, stream>>>(hf, ffn2, g2, be2, out, nullptr);
}
